// DotProductAttention_27728308863581
// MI455X (gfx1250) — matrix operand order repaired and checked by reading, not compiled, not run on hardware
//
#include <hip/hip_runtime.h>
#include <hip/hip_bf16.h>

typedef __attribute__((ext_vector_type(16))) __bf16 v16bf;
typedef __attribute__((ext_vector_type(8)))  float  v8f;
typedef __attribute__((ext_vector_type(4)))  unsigned int v4u;
typedef __attribute__((ext_vector_type(8)))  int v8i;
typedef __attribute__((ext_vector_type(4)))  int v4i;

#define WMMA_BF16(A_, B_, C_) \
  __builtin_amdgcn_wmma_f32_16x16x32_bf16(false, (A_), false, (B_), (short)0, (C_), false, false)

// ---- 32-bit LDS byte offset of a shared-memory pointer (for the TDM D#) ----
__device__ __forceinline__ unsigned lds_off(const void* p) {
  return (unsigned)(unsigned long long)(__attribute__((address_space(3))) const char*)p;
}

// ---- Tensor Data Mover: async DMA of a rows x cols f32 tile, global -> LDS ----
// D# per CDNA5 ISA ch.8: group0 = {flags/count, lds_addr, global_addr, type},
// group1 = {mask|data_size, dims, tile dims, stride}; groups 2/3 zero (2D tile).
__device__ __forceinline__ void tdm_load_f32(unsigned lds_byte, const float* gsrc,
                                             int rows, int cols) {
  const unsigned long long ga = (unsigned long long)gsrc;
  v4u g0;
  g0[0] = 1u;                                               // count=1 (valid user D#)
  g0[1] = lds_byte;                                         // LDS dest byte address
  g0[2] = (unsigned)ga;                                     // global_addr[31:0]
  g0[3] = (unsigned)((ga >> 32) & 0x1FFFFFFu) | (2u << 30); // addr[56:32] | type=2
  v8i g1;
  g1[0] = (int)(2u << 16);                                  // wg_mask=0, data_size=2 (4B)
  g1[1] = (int)((cols & 0xFFFF) << 16);                     // tensor_dim0 lo16
  g1[2] = (int)(((unsigned)cols >> 16) | ((rows & 0xFFFF) << 16)); // dim0 hi | dim1 lo
  g1[3] = (int)((cols & 0xFFFF) << 16);                     // dim1 hi(=0) | tile_dim0
  g1[4] = rows;                                             // tile_dim1 | tile_dim2=0
  g1[5] = cols;                                             // tensor_dim0_stride lo32
  g1[6] = 0;
  g1[7] = 0;
  const v4i z4 = {0, 0, 0, 0};
  const v8i z8 = {0, 0, 0, 0, 0, 0, 0, 0};
#if __has_builtin(__builtin_amdgcn_tensor_load_to_lds)
  // amdgpu-toolchain (clang-23 / therock-10.0) 6-arg form
  __builtin_amdgcn_tensor_load_to_lds(g0, g1, z4, z4, z8, 0);
#else
  asm volatile("tensor_load_to_lds %0, %1" :: "s"(g0), "s"(g1) : "memory");
#endif
}

__device__ __forceinline__ void wait_tensor0() {
#if __has_builtin(__builtin_amdgcn_s_wait_tensorcnt)
  __builtin_amdgcn_s_wait_tensorcnt(0);
#else
  asm volatile("s_wait_tensorcnt 0x0" ::: "memory");
#endif
}

// Load 16 contiguous bf16 (32B, 16B-aligned halves) from LDS as a WMMA fragment.
__device__ __forceinline__ v16bf ld_frag(const __bf16* p) {
  union { uint4 u[2]; v16bf v; } u;
  u.u[0] = *(const uint4*)(p);
  u.u[1] = *(const uint4*)(p + 8);
  return u.v;
}
// Two 8-element runs (A-matrix layout: K 0..7 / 16..23 per lane half).
__device__ __forceinline__ v16bf ld_frag2(const __bf16* p0, const __bf16* p1) {
  union { uint4 u[2]; v16bf v; } u;
  u.u[0] = *(const uint4*)(p0);
  u.u[1] = *(const uint4*)(p1);
  return u.v;
}
// Build an A-fragment from two runs of 8 f32 in global memory (cvt to bf16).
__device__ __forceinline__ v16bf cvt_frag(const float* p0, const float* p1) {
  v16bf r;
#pragma unroll
  for (int i = 0; i < 8; ++i) r[i] = (__bf16)p0[i];
#pragma unroll
  for (int i = 0; i < 8; ++i) r[8 + i] = (__bf16)p1[i];
  return r;
}

// Fused dual-softmax attention pass:
//   out[q, d] = sum_j exp(Q_q . V_j) * mQ[q] * mV[j] * V[j, d] / (rowsum + eps)
// Called twice with (Q,V) swapped to produce a_attend and b_attend.
__global__ __launch_bounds__(128, 1)
void dual_attend_kernel(const float* __restrict__ Q, const float* __restrict__ V,
                        const int* __restrict__ mQ, const int* __restrict__ mV,
                        float* __restrict__ out, int Lq, int Lv) {
  constexpr int D = 512;
  extern __shared__ char smem[];
  float*  fstage = (float*)smem;              // [32][512] raw f32 V tile (TDM destination)
  __bf16* lV     = (__bf16*)(smem + 32 * D * 4); // [32][512] bf16 row-major (score B-op)
  __bf16* lVt    = lV + 32 * D;               // [512][32] bf16 transposed (attend B-op)
  __bf16* lP     = lVt + 32 * D;              // [4 waves][16][32] probability staging

  const int bz   = blockIdx.y;                // batch
  const int row0 = blockIdx.x * 64;           // 64 Q-rows per workgroup
  const int tid  = threadIdx.x;
  const int wave = tid >> 5;
  const int lane = tid & 31;
  const int hl   = lane >> 4;                 // lane half (0/1)
  const int l16  = lane & 15;
  const int mbase = row0 + wave * 16;         // this wave's 16-row M tile

  // ---- persistent A fragments: 16 x (16x32 bf16), K covers D=512 ----
  v16bf afrag[16];
  {
    const float* qb = Q + ((size_t)bz * Lq + mbase + l16) * D;
#pragma unroll
    for (int kk = 0; kk < 16; ++kk) {
      const int d0 = kk * 32 + hl * 8;        // lanes 0-15: K 0..7|16..23 ; 16-31: 8..15|24..31
      afrag[kk] = cvt_frag(qb + d0, qb + d0 + 16);
    }
  }

  // row masks (per C-layout: VGPR v -> row v + 8*half)
  float ma[8];
#pragma unroll
  for (int v = 0; v < 8; ++v)
    ma[v] = (float)mQ[(size_t)bz * Lq + mbase + v + hl * 8];

  v8f acc[32];                                // 16 x 512 f32 output accumulator
#pragma unroll
  for (int t = 0; t < 32; ++t) acc[t] = {};
  v8f rs = {};                                // per-lane partial row sums

  // ---- prologue: each wave DMAs its 8-row slice of V tile 0 into fstage ----
  const unsigned fbase = lds_off(fstage) + wave * 8 * D * 4;
  tdm_load_f32(fbase, V + ((size_t)bz * Lv + wave * 8) * D, 8, D);

  for (int j0 = 0; j0 < Lv; j0 += 32) {
    wait_tensor0();                           // own TDM slice landed in LDS
    __syncthreads();                          // all waves' slices landed; lV/lVt free

    // ---- convert staged f32 tile -> bf16 in both orientations ----
#pragma unroll 4
    for (int rep = 0; rep < 32; ++rep) {
      const int idx = rep * 128 + tid;        // each slot = 4 elements
      const int r = idx >> 7;                 // row 0..31
      const int c = (idx & 127) * 4;          // col 0..508
      const float4 f = *(const float4*)(fstage + r * D + c);
      const __bf16 b0 = (__bf16)f.x, b1 = (__bf16)f.y, b2 = (__bf16)f.z, b3 = (__bf16)f.w;
      __bf16* row = lV + r * D + c;
      row[0] = b0; row[1] = b1; row[2] = b2; row[3] = b3;
      lVt[(c + 0) * 32 + r] = b0;
      lVt[(c + 1) * 32 + r] = b1;
      lVt[(c + 2) * 32 + r] = b2;
      lVt[(c + 3) * 32 + r] = b3;
    }
    __syncthreads();                          // fstage fully consumed, bf16 tiles ready

    // ---- kick async DMA of the NEXT tile; overlaps with WMMA phase below ----
    if (j0 + 32 < Lv)
      tdm_load_f32(fbase, V + ((size_t)bz * Lv + j0 + 32 + wave * 8) * D, 8, D);

    const float mb0 = (float)mV[(size_t)bz * Lv + j0 + l16];
    const float mb1 = (float)mV[(size_t)bz * Lv + j0 + 16 + l16];

    // ---- scores: S(16x32) = A(16x512) . Vtile(32x512)^T, 32 WMMAs ----
    v8f c0 = {}, c1 = {};
#pragma unroll
    for (int kk = 0; kk < 16; ++kk) {
      const __bf16* bp0 = lV + l16 * D        + kk * 32 + hl * 8;
      const __bf16* bp1 = lV + (16 + l16) * D + kk * 32 + hl * 8;
      const v16bf bf0 = ld_frag2(bp0, bp0 + 16);
      const v16bf bf1 = ld_frag2(bp1, bp1 + 16);
      c0 = WMMA_BF16(afrag[kk], bf0, c0);
      c1 = WMMA_BF16(afrag[kk], bf1, c1);
    }

    // ---- e = exp(S) * masks; accumulate row sums; stage P to LDS (bf16) ----
    __bf16* pw = lP + wave * 512;
#pragma unroll
    for (int v = 0; v < 8; ++v) {
      const float e0 = __expf(c0[v]) * ma[v] * mb0;
      const float e1 = __expf(c1[v]) * ma[v] * mb1;
      rs[v] += e0 + e1;
      const int m = v + hl * 8;
      pw[m * 32 + l16]      = (__bf16)e0;
      pw[m * 32 + 16 + l16] = (__bf16)e1;
    }
    asm volatile("s_wait_dscnt 0" ::: "memory"); // C-layout -> A-layout transpose via LDS

    const __bf16* pp = pw + l16 * 32 + hl * 8;
    const v16bf pf = ld_frag2(pp, pp + 16);

    // ---- attend: O(16x512) += P(16x32) . Vtile(32x512), 32 WMMAs ----
#pragma unroll
    for (int t = 0; t < 32; ++t) {
      const __bf16* vp = lVt + (t * 16 + l16) * 32 + hl * 8;
      const v16bf bf = ld_frag2(vp, vp + 16);
      acc[t] = WMMA_BF16(pf, bf, acc[t]);
    }
  }

  // ---- reduce row sums across the 16-lane halves (wave32 shuffles) ----
#pragma unroll
  for (int m = 1; m < 16; m <<= 1) {
#pragma unroll
    for (int v = 0; v < 8; ++v) rs[v] += __shfl_xor(rs[v], m, 32);
  }
  float scale[8];
#pragma unroll
  for (int v = 0; v < 8; ++v) scale[v] = 1.0f / (rs[v] + 1e-7f);

  // ---- normalize + store (C-layout: lane = d column, VGPR = row) ----
  float* ob = out + ((size_t)bz * Lq + mbase) * D;
#pragma unroll
  for (int t = 0; t < 32; ++t) {
#pragma unroll
    for (int v = 0; v < 8; ++v) {
      ob[(size_t)(v + hl * 8) * D + t * 16 + l16] = acc[t][v] * scale[v];
    }
  }
}

extern "C" void kernel_launch(void* const* d_in, const int* in_sizes, int n_in,
                              void* d_out, int out_size, void* d_ws, size_t ws_size,
                              hipStream_t stream) {
  const float* A  = (const float*)d_in[0];   // [64,1024,512] f32
  const float* Bv = (const float*)d_in[1];   // [64,1024,512] f32
  const int* maskA = (const int*)d_in[2];    // [64,1024]
  const int* maskB = (const int*)d_in[3];    // [64,1024]
  float* out = (float*)d_out;

  constexpr int Bn = 64, La = 1024, Lb = 1024, D = 512;
  const size_t outA = (size_t)Bn * La * D;

  const dim3 block(128);
  // fstage (64KB f32) + lV (32KB) + lVt (32KB) + lP (4KB) = 132 KB dynamic LDS
  const size_t shmem = (size_t)32 * D * 4 + ((size_t)32 * D + (size_t)D * 32 + 4 * 16 * 32) * 2;

  // a_attend = softmax_j(e) . B   -> out[0 .. outA)
  dim3 gridA(La / 64, Bn);
  dual_attend_kernel<<<gridA, block, shmem, stream>>>(A, Bv, maskA, maskB, out, La, Lb);

  // b_attend = softmax_i(e)^T . A -> out[outA .. 2*outA)  (scores are symmetric)
  dim3 gridB(Lb / 64, Bn);
  dual_attend_kernel<<<gridB, block, shmem, stream>>>(Bv, A, maskB, maskA, out + outA, Lb, La);
}
